// CrfDecoderABC_45801531244729
// MI455X (gfx1250) — matrix operand order repaired and checked by reading, not compiled, not run on hardware
//
#include <hip/hip_runtime.h>
#include <hip/hip_fp16.h>
#include <math.h>

typedef _Float16 v16h __attribute__((ext_vector_type(16)));
typedef _Float16 v8h  __attribute__((ext_vector_type(8)));
typedef float    v8f  __attribute__((ext_vector_type(8)));
typedef float    v4f  __attribute__((ext_vector_type(4)));

namespace {
constexpr int kB = 64, kT = 1024, kC = 2, kN = 128;
constexpr int kRows = 16;    // batch rows per workgroup == WMMA M
constexpr int kPStr = 144;   // padded f32 row stride: 144 % 64 == 16 banks, 576 B rows (16-B mult)
constexpr int kHStr = 136;   // padded f16 row stride: 272 B rows (16-B mult), aligned 8-half chunks
}

__global__ __launch_bounds__(256) void crf_logz_scan(
    const float* __restrict__ emissions,      // [B,T,C,N]
    const int*   __restrict__ lengths,        // [B]
    const float* __restrict__ transitions,    // [1,C,N,N]
    const float* __restrict__ start_trans,    // [1,C,N]
    const float* __restrict__ end_trans,      // [1,C,N]
    float*       __restrict__ out)            // [B,C]
{
    __shared__ __align__(16) float     praw[kRows][kPStr]; // exp-domain alpha (raw scale)
    __shared__ __align__(16) _Float16  Pf16[kRows][kHStr]; // normalized f16 copy (WMMA A src)

    const int tid  = threadIdx.x;
    const int wave = tid >> 5;            // 0..7 : owns output columns [16*wave, 16*wave+16)
    const int lane = tid & 31;
    const int hi   = lane >> 4;
    const int l16  = lane & 15;
    const int c    = blockIdx.y;          // channel
    const int b0   = blockIdx.x * kRows;  // batch base

    const int n = wave * 16 + l16;        // column of D this lane holds

    // ---- B fragments: E = exp(transitions[c]) in WMMA B layout, built once ----
    // B 32x16 f16 layout: lane = N column; halves e=0..15 map to K = hi*8 + (e&7) + (e>>3)*16.
    v16h Bf[4];
    {
        const float* tr = transitions + (size_t)c * kN * kN;
        #pragma unroll
        for (int kk = 0; kk < 4; ++kk) {
            #pragma unroll
            for (int e = 0; e < 16; ++e) {
                const int k = kk * 32 + hi * 8 + (e & 7) + (e >> 3) * 16;
                Bf[kk][e] = (_Float16)__expf(tr[k * kN + n]);
            }
        }
    }

    // ---- per-lane info for the 8 output rows held in the D accumulator ----
    const float* ebase[8];
    int len8[8];
    #pragma unroll
    for (int i = 0; i < 8; ++i) {
        const int row = i + 8 * hi;                       // D VGPR i -> matrix row
        const int bb  = b0 + row;
        ebase[i] = emissions + (size_t)bb * (kT * kC * kN) + (size_t)c * kN + n;
        len8[i]  = lengths[bb];
    }

    // block-uniform scan bound: all rows frozen once t >= max(lengths in tile)
    int tEnd = 0;
    #pragma unroll
    for (int i = 0; i < kRows; ++i) tEnd = max(tEnd, lengths[b0 + i]);

    // ---- reduction view: thread (r,q) owns row r, contiguous cols [8q, 8q+8) ----
    const int r = tid >> 4, q = tid & 15;
    const int col0 = q * 8;
    float mval = 0.f;                                     // log-offset of row r (q==0 lanes)

    // ---- init: alpha0 = start + emissions[:,0]; praw = exp(alpha0 - rowmax) ----
    {
        const float* em0 = emissions + (size_t)(b0 + r) * (kT * kC * kN) + (size_t)c * kN;
        float a8[8], mx = -INFINITY;
        #pragma unroll
        for (int j = 0; j < 8; ++j) {
            a8[j] = start_trans[c * kN + col0 + j] + em0[col0 + j];
            mx = fmaxf(mx, a8[j]);
        }
        #pragma unroll
        for (int s = 1; s < 16; s <<= 1) mx = fmaxf(mx, __shfl_xor(mx, s, 16));
        mval = mx;
        #pragma unroll
        for (int j = 0; j < 8; ++j) praw[r][col0 + j] = __expf(a8[j] - mx);
    }
    __syncthreads();

    for (int t = 1; t < tEnd; ++t) {
        // prefetch this step's emission tile (latency hidden under phases below)
        float em[8];
        #pragma unroll
        for (int i = 0; i < 8; ++i) em[i] = ebase[i][(size_t)t * (kC * kN)];

        // phase 1: renormalize row (one log per row), write packed f16 operand copy
        {
            v4f a0 = *(const v4f*)&praw[r][col0];          // ds_load_b128
            v4f a1 = *(const v4f*)&praw[r][col0 + 4];      // ds_load_b128
            float mx = 0.f;
            #pragma unroll
            for (int j = 0; j < 4; ++j) mx = fmaxf(mx, fmaxf(a0[j], a1[j]));
            #pragma unroll
            for (int s = 1; s < 16; s <<= 1) mx = fmaxf(mx, __shfl_xor(mx, s, 16));
            const float inv = __builtin_amdgcn_rcpf(mx);
            if (q == 0) mval += __logf(mx);
            v8h ph;
            #pragma unroll
            for (int j = 0; j < 4; ++j) {
                a0[j] *= inv;  a1[j] *= inv;
                ph[j] = (_Float16)a0[j];  ph[j + 4] = (_Float16)a1[j];
            }
            *(v4f*)&praw[r][col0]     = a0;                // ds_store_b128
            *(v4f*)&praw[r][col0 + 4] = a1;                // ds_store_b128
            *(v8h*)&Pf16[r][col0]     = ph;                // ds_store_b128
        }
        __syncthreads();

        // phase 2: D(16x16,f32) = P(16x128,f16) x E(128x16,f16), 4 chained WMMAs.
        // A 16x32 layout: lane = row; chunk0 halves = K kk*32+hi*8.., chunk1 = +16.
        union { v16h v; v8h h[2]; } Af[4];
        const _Float16* prow = &Pf16[l16][0];
        #pragma unroll
        for (int kk = 0; kk < 4; ++kk) {
            Af[kk].h[0] = *(const v8h*)(prow + kk * 32 + hi * 8);        // ds_load_b128
            Af[kk].h[1] = *(const v8h*)(prow + kk * 32 + 16 + hi * 8);   // ds_load_b128
        }
        v8f acc = {};
        #pragma unroll
        for (int kk = 0; kk < 4; ++kk)
            acc = __builtin_amdgcn_wmma_f32_16x16x32_f16(
                false, Af[kk].v, false, Bf[kk], (short)0, acc, false, false);

        // phase 3: fold emissions in exp domain; ragged-length mask keeps old row
        #pragma unroll
        for (int i = 0; i < 8; ++i) {
            const int row   = i + 8 * hi;
            const float old = praw[row][n];
            const float nv  = acc[i] * __expf(em[i]);
            praw[row][n] = (t < len8[i]) ? nv : old;
        }
        __syncthreads();
    }

    // ---- finalize: logZ[b,c] = mval + log( sum_n praw[n] * exp(end[n]) ) ----
    {
        float s = 0.f;
        #pragma unroll
        for (int j = 0; j < 8; ++j)
            s += praw[r][col0 + j] * __expf(end_trans[c * kN + col0 + j]);
        #pragma unroll
        for (int sh = 1; sh < 16; sh <<= 1) s += __shfl_xor(s, sh, 16);
        if (q == 0) out[(b0 + r) * kC + c] = mval + __logf(s);
    }
}

extern "C" void kernel_launch(void* const* d_in, const int* in_sizes, int n_in,
                              void* d_out, int out_size, void* d_ws, size_t ws_size,
                              hipStream_t stream) {
    (void)in_sizes; (void)n_in; (void)d_ws; (void)ws_size; (void)out_size;
    const float* emissions = (const float*)d_in[0];
    const int*   lengths   = (const int*)d_in[1];
    const float* trans     = (const float*)d_in[2];
    const float* start_tr  = (const float*)d_in[3];
    const float* end_tr    = (const float*)d_in[4];
    float* out = (float*)d_out;

    dim3 grid(kB / kRows, kC);   // 4 x 2 = 8 workgroups, each owns 16 batch rows of one channel
    crf_logz_scan<<<grid, 256, 0, stream>>>(emissions, lengths, trans, start_tr, end_tr, out);
}
